// GroupedMultiQueryAttention_37873021616549
// MI455X (gfx1250) — compile-verified, matrix operand order checked by reading, not run on hardware
//
#include <hip/hip_runtime.h>

// ---------------------------------------------------------------------------
// GroupedMultiQueryAttention for MI455X (gfx1250, wave32, WMMA)
// Key algebraic simplification: the reference einsum sums the kv-head axis g,
// and RoPE is linear in k, so k/v collapse to single [N,64] tensors whose
// projection weights are the g-sum of w_kv columns.
// All GEMMs run on v_wmma_f32_16x16x32_bf16 (f32 accumulate).
// Attention stages K/V blocks into LDS with global_load_async_to_lds_b128
// (ASYNCcnt) shared by all 4 waves of the block.
// ---------------------------------------------------------------------------

typedef __attribute__((ext_vector_type(16))) __bf16        v16bf;
typedef __attribute__((ext_vector_type(8)))  float         v8f;
typedef __attribute__((ext_vector_type(4)))  unsigned int  v4u;

#define B_SZ   2
#define N_SEQ  2048
#define C_DIM  1024
#define N_HEAD 16
#define HD     64

__device__ __forceinline__ v8f wmma_bf16(v16bf a, v16bf b, v8f c) {
  return __builtin_amdgcn_wmma_f32_16x16x32_bf16(
      /*neg_a=*/false, a, /*neg_b=*/false, b,
      /*c_mod=*/(short)0, c, /*reuse_a=*/false, /*reuse_b=*/false);
}

// Load a 16x32 bf16 fragment in the CDNA5 WMMA A/B register layout.
// Memory layout: element (row, k) at p0[row*ldk + k]  (k contiguous).
// ISA 7.12.2 (16-bit A 16x32): lanes 0-15 hold K {0..7,16..23},
// lanes 16-31 hold K {8..15,24..31}; 2 bf16 per VGPR -> two 16B loads.
__device__ __forceinline__ v16bf load_frag(const __bf16* p0, int ldk, int lane) {
  const __bf16* p = p0 + (size_t)(lane & 15) * (size_t)ldk + ((lane & 16) ? 8 : 0);
  union { v4u u[2]; v16bf v; } f;
  f.u[0] = *(const v4u*)(p);        // K = ko .. ko+7   -> VGPR 0..3
  f.u[1] = *(const v4u*)(p + 16);   // K = ko+16..ko+23 -> VGPR 4..7
  return f.v;
}

// Async copy of one 16-byte chunk from global to LDS (per-lane addresses).
// Tracked on ASYNCcnt; fence with s_wait_asynccnt + block barrier.
__device__ __forceinline__ void async_b128(const __bf16* gaddr, __bf16* lds) {
  unsigned laddr = (unsigned)(size_t)lds;   // low 32 bits of shared aperture = LDS offset
  asm volatile("global_load_async_to_lds_b128 %0, %1, off"
               :: "v"(laddr), "v"(gaddr) : "memory");
}

// Row reductions across the 16 lanes of a half-wave (WMMA C-layout rows).
__device__ __forceinline__ float rowmax16(float x) {
  x = fmaxf(x, __shfl_xor(x, 1, 32));
  x = fmaxf(x, __shfl_xor(x, 2, 32));
  x = fmaxf(x, __shfl_xor(x, 4, 32));
  x = fmaxf(x, __shfl_xor(x, 8, 32));
  return x;
}
__device__ __forceinline__ float rowsum16(float x) {
  x += __shfl_xor(x, 1, 32);
  x += __shfl_xor(x, 2, 32);
  x += __shfl_xor(x, 4, 32);
  x += __shfl_xor(x, 8, 32);
  return x;
}

// ---------------------------------------------------------------------------
// Packing kernels (memory-bound, trivial)
// ---------------------------------------------------------------------------
__global__ void pack_f32_to_bf16(const float* __restrict__ src,
                                 __bf16* __restrict__ dst, int n) {
  int i = blockIdx.x * blockDim.x + threadIdx.x;
  if (i < n) dst[i] = (__bf16)src[i];
}

// dst[o*1024 + c] = src[c*1024 + o]  (weights stored as [n_out][K])
__global__ void pack_transpose1024(const float* __restrict__ src,
                                   __bf16* __restrict__ dst) {
  int i = blockIdx.x * blockDim.x + threadIdx.x;   // 1024*1024 threads
  int c = i & 1023;
  int o = i >> 10;
  dst[(size_t)o * 1024 + c] = (__bf16)src[(size_t)c * 1024 + o];
}

// g-reduced kv projection weights, transposed: wkT/wvT[o*1024 + c]
__global__ void pack_wkv(const float* __restrict__ w_kv,
                         __bf16* __restrict__ wkT,
                         __bf16* __restrict__ wvT) {
  int i = blockIdx.x * blockDim.x + threadIdx.x;   // 64*1024 threads
  if (i >= 64 * 1024) return;
  int c = i & 1023;
  int o = i >> 10;                                 // 0..63
  const float* row = w_kv + (size_t)c * 512;
  float sk = row[o] + row[64 + o] + row[128 + o] + row[192 + o];
  float sv = row[256 + o] + row[320 + o] + row[384 + o] + row[448 + o];
  wkT[(size_t)o * 1024 + c] = (__bf16)sk;
  wvT[(size_t)o * 1024 + c] = (__bf16)sv;
}

// ---------------------------------------------------------------------------
// Generic per-wave 16x16-tile bf16 GEMM:  out = A[M,K] * Bt[Nout,K]^T
// grid = (Nout/64, M/16), block = 128 (4 waves, each one 16-col tile)
// mode 0: Q proj   -> Qb[((b*16+h)*N + n)*64 + d]        (bf16)
// mode 1: k_sum    -> Kb[(b*N + n)*64 + d]               (bf16)
// mode 2: v_sum^T  -> VbT[(b*64 + d)*N + n]              (bf16)
// mode 3: out proj -> out[row*1024 + col]                (f32)
// ---------------------------------------------------------------------------
__global__ __launch_bounds__(128) void gemm16(const __bf16* __restrict__ A,
                                              const __bf16* __restrict__ Bt,
                                              void* __restrict__ outp,
                                              int K, int mode) {
  int lane = threadIdx.x & 31;
  int w    = threadIdx.x >> 5;
  int tn   = blockIdx.x * 4 + w;
  int tm   = blockIdx.y;

  const __bf16* aBase = A  + (size_t)tm * 16 * (size_t)K;
  const __bf16* bBase = Bt + (size_t)tn * 16 * (size_t)K;

  v8f c = {};
#pragma unroll 4
  for (int kk = 0; kk < K; kk += 32) {
    v16bf a = load_frag(aBase + kk, K, lane);
    v16bf b = load_frag(bBase + kk, K, lane);
    c = wmma_bf16(a, b, c);
  }

  int half = lane >> 4;
  int col  = tn * 16 + (lane & 15);
#pragma unroll
  for (int r = 0; r < 8; r++) {
    int row = tm * 16 + r + 8 * half;
    float v = c[r];
    if (mode == 0) {
      int bb = row / N_SEQ, n = row % N_SEQ;
      int hh = col >> 6,    d = col & 63;
      ((__bf16*)outp)[(((size_t)(bb * N_HEAD + hh) * N_SEQ) + n) * HD + d] = (__bf16)v;
    } else if (mode == 1) {
      ((__bf16*)outp)[(size_t)row * HD + col] = (__bf16)v;
    } else if (mode == 2) {
      int bb = row / N_SEQ, n = row % N_SEQ;
      ((__bf16*)outp)[((size_t)(bb * HD + col)) * N_SEQ + n] = (__bf16)v;
    } else {
      ((float*)outp)[(size_t)row * 1024 + col] = v;
    }
  }
}

// ---------------------------------------------------------------------------
// RoPE in-place on bf16 buffer laid out [R][N][64]; thread per (r, n, d<32)
// ---------------------------------------------------------------------------
__global__ void rope_kernel(__bf16* __restrict__ buf, int R) {
  int i = blockIdx.x * blockDim.x + threadIdx.x;
  int total = R * N_SEQ * 32;
  if (i >= total) return;
  int d = i & 31;
  int t = i >> 5;
  int n = t % N_SEQ;
  int r = t / N_SEQ;
  // inv_freq = 10000^(-2d/64) = 2^(-(2d/64) * log2(10000))
  float inv = exp2f(-(float)(2 * d) * (1.0f / 64.0f) * 13.287712379549449f);
  float f = (float)n * inv;
  float s, c;
  __sincosf(f, &s, &c);
  size_t base = ((size_t)r * N_SEQ + n) * HD + d;
  float q0 = (float)buf[base];
  float q1 = (float)buf[base + 32];
  buf[base]      = (__bf16)(q0 * c - q1 * s);
  buf[base + 32] = (__bf16)(q1 * c + q0 * s);
}

// ---------------------------------------------------------------------------
// Flash-style attention. Block = 128 threads (4 waves); all waves share one
// (b,h); each wave owns one 16-row tile. Per 32-col j-block:
//   1) block cooperatively async-stages K(32x64) and V^T(64x32) into LDS
//      (global_load_async_to_lds_b128, ASYNCcnt), fence + barrier;
//   2) 4 WMMAs for S from LDS;
//   3) online softmax (shfl half-wave reductions), P bounced C-layout ->
//      A-layout through a wave-private LDS slab (s_wait_dscnt);
//   4) 4 WMMAs for P @ V^T from LDS.
// grid = (N/64, H, B)
// ---------------------------------------------------------------------------
__global__ __launch_bounds__(128) void attn_kernel(const __bf16* __restrict__ Qb,
                                                   const __bf16* __restrict__ Kb,
                                                   const __bf16* __restrict__ VbT,
                                                   __bf16* __restrict__ Ob) {
  __shared__ alignas(16) __bf16 kTile[32 * 64];      // 4 KiB: rows j..j+31, k=0..63
  __shared__ alignas(16) __bf16 vTile[64 * 32];      // 4 KiB: rows d=0..63, jj=0..31
  __shared__ alignas(16) __bf16 pbuf[4][16 * 32];    // 4 KiB: per-wave P slab

  int tid  = threadIdx.x;
  int lane = tid & 31;
  int w    = tid >> 5;
  int half = lane >> 4;
  int rowTile = blockIdx.x * 4 + w;
  int h = blockIdx.y;
  int b = blockIdx.z;

  const __bf16* Qbase = Qb + (((size_t)(b * N_HEAD + h) * N_SEQ) + (size_t)rowTile * 16) * HD;
  const __bf16* Kbase = Kb + (size_t)b * N_SEQ * HD;
  const __bf16* Vbase = VbT + (size_t)b * HD * N_SEQ;

  // Q row tile is fixed for this wave: preload both K-halves of the A frag.
  v16bf aQ0 = load_frag(Qbase, HD, lane);        // k = 0..31
  v16bf aQ1 = load_frag(Qbase + 32, HD, lane);   // k = 32..63

  v8f o0 = {}, o1 = {}, o2 = {}, o3 = {};
  float mr[8], lr[8];
#pragma unroll
  for (int r = 0; r < 8; r++) { mr[r] = -1e30f; lr[r] = 0.0f; }

  const float scale = 0.125f;   // D^-0.5, D = 64
  __bf16* pb = &pbuf[w][0];

  for (int j = 0; j < N_SEQ; j += 32) {
    // ---- 1) async-stage K and V^T blocks into LDS (2 KiB per issue) ----
    {
      int c0 = tid, c1 = tid + 128;        // 256 x 16B chunks each for K and V
      // K block: chunk c -> row j + c/8, elems (c&7)*8..+7
      async_b128(Kbase + (size_t)(j + (c0 >> 3)) * HD + (c0 & 7) * 8, &kTile[c0 * 8]);
      async_b128(Kbase + (size_t)(j + (c1 >> 3)) * HD + (c1 & 7) * 8, &kTile[c1 * 8]);
      // V^T block: chunk c -> row d = c/4, elems j + (c&3)*8..+7
      async_b128(Vbase + (size_t)(c0 >> 2) * N_SEQ + j + (c0 & 3) * 8, &vTile[c0 * 8]);
      async_b128(Vbase + (size_t)(c1 >> 2) * N_SEQ + j + (c1 & 3) * 8, &vTile[c1 * 8]);
    }
    asm volatile("s_wait_asynccnt 0x0" ::: "memory");
    __syncthreads();

    // ---- 2) S tiles for columns j..j+15 and j+16..j+31 (from LDS) ----
    v8f s0 = {}, s1 = {};
    s0 = wmma_bf16(aQ0, load_frag(kTile,                 HD, lane), s0);
    s0 = wmma_bf16(aQ1, load_frag(kTile + 32,            HD, lane), s0);
    s1 = wmma_bf16(aQ0, load_frag(kTile + 16 * HD,       HD, lane), s1);
    s1 = wmma_bf16(aQ1, load_frag(kTile + 16 * HD + 32,  HD, lane), s1);

    // ---- 3) online softmax update, per C-layout row (r + 8*half) ----
#pragma unroll
    for (int r = 0; r < 8; r++) {
      float x0 = s0[r] * scale;
      float x1 = s1[r] * scale;
      float mx   = rowmax16(fmaxf(x0, x1));
      float mnew = fmaxf(mr[r], mx);
      float alpha = __expf(mr[r] - mnew);
      float p0 = __expf(x0 - mnew);
      float p1 = __expf(x1 - mnew);
      float rs = rowsum16(p0 + p1);
      lr[r] = lr[r] * alpha + rs;
      mr[r] = mnew;
      o0[r] *= alpha; o1[r] *= alpha; o2[r] *= alpha; o3[r] *= alpha;
      int row = r + 8 * half;
      pb[row * 32 + (lane & 15)]      = (__bf16)p0;
      pb[row * 32 + 16 + (lane & 15)] = (__bf16)p1;
    }

    // C-layout -> A-layout transpose bounce through LDS (wave-private slab)
    asm volatile("s_wait_dscnt 0" ::: "memory");
    v16bf pf = load_frag(pb, 32, lane);

    // ---- 4) O(16x64) += P(16x32) @ V^T tiles (from LDS) ----
    o0 = wmma_bf16(pf, load_frag(vTile + 0 * 16 * 32, 32, lane), o0);
    o1 = wmma_bf16(pf, load_frag(vTile + 1 * 16 * 32, 32, lane), o1);
    o2 = wmma_bf16(pf, load_frag(vTile + 2 * 16 * 32, 32, lane), o2);
    o3 = wmma_bf16(pf, load_frag(vTile + 3 * 16 * 32, 32, lane), o3);

    __syncthreads();   // all reads done before next iteration's async writes
  }

  // Normalize and store O in [B][N][H*64] bf16 (A-operand layout for out-proj)
  int nc = lane & 15;
#pragma unroll
  for (int r = 0; r < 8; r++) {
    int row = rowTile * 16 + r + 8 * half;
    float invl = 1.0f / lr[r];
    size_t obase = ((size_t)b * N_SEQ + row) * (size_t)(N_HEAD * HD) + (size_t)h * HD;
    Ob[obase + 0  + nc] = (__bf16)(o0[r] * invl);
    Ob[obase + 16 + nc] = (__bf16)(o1[r] * invl);
    Ob[obase + 32 + nc] = (__bf16)(o2[r] * invl);
    Ob[obase + 48 + nc] = (__bf16)(o3[r] * invl);
  }
}

// ---------------------------------------------------------------------------
// Host-side orchestration
// ---------------------------------------------------------------------------
extern "C" void kernel_launch(void* const* d_in, const int* in_sizes, int n_in,
                              void* d_out, int out_size, void* d_ws, size_t ws_size,
                              hipStream_t stream) {
  const float* x     = (const float*)d_in[0];   // [2,2048,1024]
  const float* w_q   = (const float*)d_in[1];   // [1024,1024]
  const float* w_kv  = (const float*)d_in[2];   // [1024,512]
  const float* w_out = (const float*)d_in[3];   // [1024,1024]

  char* ws = (char*)d_ws;
  __bf16* xb  = (__bf16*)(ws);             // 4096*1024        ( 8 MiB)
  __bf16* wqT = (__bf16*)(ws + 8388608);   // 1024*1024        ( 2 MiB)
  __bf16* wkT = (__bf16*)(ws + 10485760);  // 64*1024          (128 KiB)
  __bf16* wvT = (__bf16*)(ws + 10616832);  // 64*1024          (128 KiB)
  __bf16* woT = (__bf16*)(ws + 10747904);  // 1024*1024        ( 2 MiB)
  __bf16* Qb  = (__bf16*)(ws + 12845056);  // 2*16*2048*64     ( 8 MiB)
  __bf16* Kb  = (__bf16*)(ws + 21233664);  // 2*2048*64        (512 KiB)
  __bf16* VbT = (__bf16*)(ws + 21757952);  // 2*64*2048        (512 KiB)
  __bf16* Ob  = (__bf16*)(ws + 22282240);  // 4096*1024        ( 8 MiB)

  // 1) pack to bf16 (x as-is; weights transposed; w_kv g-reduced)
  pack_f32_to_bf16<<<16384, 256, 0, stream>>>(x, xb, 4194304);
  pack_transpose1024<<<4096, 256, 0, stream>>>(w_q, wqT);
  pack_wkv<<<256, 256, 0, stream>>>(w_kv, wkT, wvT);
  pack_transpose1024<<<4096, 256, 0, stream>>>(w_out, woT);

  // 2) projections (M = B*N = 4096 rows)
  gemm16<<<dim3(16, 256), 128, 0, stream>>>(xb, wqT, Qb, 1024, 0);   // Q
  gemm16<<<dim3(1, 256),  128, 0, stream>>>(xb, wkT, Kb, 1024, 1);   // k_sum
  gemm16<<<dim3(1, 256),  128, 0, stream>>>(xb, wvT, VbT, 1024, 2);  // v_sum^T

  // 3) RoPE on Q (R = B*H = 32) and k_sum (R = B = 2)
  rope_kernel<<<2097152 / 256, 256, 0, stream>>>(Qb, 32);
  rope_kernel<<<131072 / 256, 256, 0, stream>>>(Kb, 2);

  // 4) attention
  attn_kernel<<<dim3(N_SEQ / 64, N_HEAD, B_SZ), 128, 0, stream>>>(Qb, Kb, VbT, Ob);

  // 5) out projection -> f32 d_out [2,2048,1024]
  gemm16<<<dim3(16, 256), 128, 0, stream>>>(Ob, woT, d_out, 1024, 3);
}
